// VanilaLSTM_54881092108739
// MI455X (gfx1250) — matrix operand order repaired and checked by reading, not compiled, not run on hardware
//
#include <hip/hip_runtime.h>
#include <hip/hip_fp16.h>

// ---------------- problem constants ----------------
#define LSTM_B   64
#define LSTM_T   2048
#define LSTM_I   256
#define LSTM_H   256
#define LSTM_O   128
#define LSTM_KT  512          // I + H concatenated K dimension
#define LSTM_4H  1024
#define NWG      32           // 16 workgroups per layer

typedef __attribute__((ext_vector_type(16))) _Float16 v16h;
typedef __attribute__((ext_vector_type(8)))  _Float16 v8h;
typedef __attribute__((ext_vector_type(8)))  float    v8f;

__device__ __forceinline__ v8f wmma_f16(v16h a, v16h b, v8f c) {
  // D = A(16x32 f16) * B(32x16 f16) + C(16x16 f32)
  return __builtin_amdgcn_wmma_f32_16x16x32_f16(
      /*neg_a=*/false, a, /*neg_b=*/false, b,
      /*c_mod=*/(short)0, c, /*reuse_a=*/false, /*reuse_b=*/false);
}

__device__ __forceinline__ float sigf(float x) {
  return 1.0f / (1.0f + __expf(-x));
}
__device__ __forceinline__ float tanh_fast(float x) {
  float e = __expf(2.0f * x);
  return (e - 1.0f) / (e + 1.0f);
}

// A-fragment (16x32, f16, contiguous layout): lane L holds row M=L%16,
// K = kb + (L<16?0:16) + {0..15}
__device__ __forceinline__ v16h load_a_f16(const _Float16* __restrict__ base,
                                           int row, int pitch, int kb, int lane) {
  const int aoff = (lane < 16) ? 0 : 16;
  const _Float16* p = base + (size_t)row * pitch + kb + aoff;
  v16h a;
  *(v8h*)&a       = *(const v8h*)p;
  *((v8h*)&a + 1) = *(const v8h*)(p + 8);
  return a;
}

__device__ __forceinline__ v16h load_a_f32(const float* __restrict__ p8,
                                           const float* __restrict__ p8b) {
  v16h a;
#pragma unroll
  for (int j = 0; j < 8; ++j) {
    a[j]     = (_Float16)p8[j];
    a[8 + j] = (_Float16)p8b[j];
  }
  return a;
}

// ---------------- weight conversion: fp32 -> f16 Wcat[l][n(=g*256+h)][k(=x||h)] ----------------
__global__ void lstm_convert_weights(const float* __restrict__ Wx,   // [2][4][256][256]
                                     const float* __restrict__ Wh,   // [2][4][256][256]
                                     _Float16* __restrict__ wcat) {  // [2][1024][512]
  const int total = 2 * LSTM_4H * LSTM_KT;
  for (int idx = blockIdx.x * 256 + threadIdx.x; idx < total; idx += gridDim.x * 256) {
    int l   = idx / (LSTM_4H * LSTM_KT);
    int rem = idx - l * (LSTM_4H * LSTM_KT);
    int n   = rem >> 9;           // 0..1023
    int kk  = rem & 511;          // 0..511
    int g   = n >> 8;             // gate
    int hr  = n & 255;            // output row within gate
    size_t wbase = (((size_t)l * 4 + g) * 256 + hr) * 256;
    float v = (kk < 256) ? Wx[wbase + kk] : Wh[wbase + (kk - 256)];
    wcat[idx] = (_Float16)v;
  }
}

// ---------------- input conversion+transpose: x[b][t][i] fp32 -> x16[t][b][i] f16 ----------------
// Streaming, fully parallel, off the recurrent critical path.
__global__ void lstm_convert_x(const float* __restrict__ x,    // [64][2048][256]
                               _Float16* __restrict__ x16) {   // [2048][64][256]
  const int totalPairs = LSTM_B * LSTM_T * (LSTM_I / 2);       // 16.8M half2 pairs
  for (int p = blockIdx.x * 256 + threadIdx.x; p < totalPairs; p += gridDim.x * 256) {
    int i2 = p & (LSTM_I / 2 - 1);            // pair index within row
    int t  = (p >> 7) & (LSTM_T - 1);
    int b  = p >> 18;
    const float* src = x + (((size_t)b * LSTM_T + t) * LSTM_I) + i2 * 2;
    _Float16* dst = x16 + (((size_t)t * LSTM_B + b) * LSTM_I) + i2 * 2;
    dst[0] = (_Float16)src[0];
    dst[1] = (_Float16)src[1];
  }
}

// ---------------- state init: zero h double-buffers + barrier counter ----------------
__global__ void lstm_init_state(_Float16* __restrict__ hbufs, unsigned* __restrict__ counter) {
  for (int i = threadIdx.x; i < 4 * LSTM_B * LSTM_H; i += 256)
    hbufs[i] = (_Float16)0.0f;
  if (threadIdx.x == 0) *counter = 0u;
}

// ---------------- persistent pipelined recurrence ----------------
// Grid: 32 wgs x 256 threads. wg = layer*16 + h-chunk. Interval k: layer0 does step k,
// layer1 does step k-1 (wavefront). Global barrier (atomic counter) between intervals.
// x-half weights (K 0..255) live in LDS; h-half weights (K 256..511) live in VGPRs
// for the whole kernel so the barrier-dependent GEMM half is pure A-load -> v_wmma.
__global__ void __launch_bounds__(256)
lstm_recurrent(const float* __restrict__ x,        // [64][2048][256] (fallback path)
               const _Float16* __restrict__ x16,   // [2048][64][256] or nullptr
               const float* __restrict__ bh,       // [2][4][256]
               const _Float16* __restrict__ wcat,  // [2][1024][512]
               _Float16* __restrict__ h0buf,       // [2][64][256] double buffer, layer0 out
               _Float16* __restrict__ h1buf,       // [2][64][256] double buffer, layer1 out
               unsigned* __restrict__ counter) {
  __shared__ _Float16 WldsX[64][LSTM_I];     // x-half weights: 64 rows x 256 K (32 KB)
  __shared__ float    gatesL[4][LSTM_B][16]; // raw gates for this h-chunk      (16 KB)
  __shared__ float    cst[LSTM_B][16];       // persistent cell state            (4 KB)
  __shared__ float    biasL[4][16];

  const int wg    = blockIdx.x;
  const int layer = wg >> 4;
  const int hc    = wg & 15;
  const int tid   = threadIdx.x;
  const int lane  = tid & 31;
  const int wave  = tid >> 5;

  // ---- preload x-half weight slice into LDS (rows g*256 + hc*16 + r, K 0..255) ----
#pragma unroll
  for (int it = 0; it < 8; ++it) {
    int idx = it * 256 + tid;          // 2048 uint4 total (64 rows x 32 uint4)
    int r   = idx >> 5;                // local row 0..63
    int cq  = idx & 31;                // uint4 within row (32 per row)
    int g   = r >> 4, rr = r & 15;
    size_t grow = ((size_t)layer * LSTM_4H + g * 256 + hc * 16 + rr);
    ((uint4*)&WldsX[r][0])[cq] = ((const uint4*)(wcat + grow * LSTM_KT))[cq];
  }
  if (tid < 64) {
    int g = tid >> 4, j = tid & 15;
    biasL[g][j] = bh[((layer * 4 + g) * 256) + hc * 16 + j];
  }
  for (int i = tid; i < LSTM_B * 16; i += 256) cst[i >> 4][i & 15] = 0.0f;

  const int m  = wave & 3;                  // M tile (batch rows)
  const int gp = wave >> 2;                 // gate pair (0:{f,i}, 1:{o,chat})
  const int arow = m * 16 + (lane & 15);
  const int na = (2 * gp) * 16 + (lane & 15);   // local LDS row for gate 2gp
  const int boff = (lane < 16) ? 0 : 16;
  const int aoff = (lane < 16) ? 0 : 16;

  // ---- preload h-half weight fragments into registers (persistent, 128 VGPRs) ----
  v16h bregA[8], bregB[8];
  {
    const _Float16* wrowA = wcat +
        (((size_t)layer * LSTM_4H + (2 * gp) * 256 + hc * 16 + (lane & 15)) * LSTM_KT) +
        LSTM_I + boff;
    const _Float16* wrowB = wrowA + (size_t)256 * LSTM_KT;  // next gate is +256 rows
#pragma unroll
    for (int ks = 0; ks < 8; ++ks) {
      bregA[ks] = *(const v16h*)(wrowA + ks * 32);
      bregB[ks] = *(const v16h*)(wrowB + ks * 32);
    }
  }
  __syncthreads();

  for (int k = 0; k <= LSTM_T; ++k) {
    const bool active = (layer == 0) ? (k < LSTM_T) : (k >= 1);
    const int  s      = (layer == 0) ? k : (k - 1);
    if (active) {
      const _Float16* hself = (layer ? h1buf : h0buf) + ((s + 1) & 1) * (LSTM_B * LSTM_H);
      const _Float16* h0in  = h0buf + (s & 1) * (LSTM_B * LSTM_H);

      v8f acc0 = {}; v8f acc1 = {};

      // ---- x-half: A from input (x/x16 for layer0, h0 f16 for layer1), B from LDS ----
      if (layer == 0) {
        if (x16) {
          const _Float16* xp = x16 + (size_t)s * (LSTM_B * LSTM_I);
          // prefetch next step's contiguous 32KB A-panel while this step computes
          if (s + 1 < LSTM_T)
            __builtin_prefetch((const char*)(xp + LSTM_B * LSTM_I) + tid * 128, 0, 0);
#pragma unroll
          for (int ks = 0; ks < 8; ++ks) {
            v16h a  = load_a_f16(xp, arow, LSTM_I, ks * 32, lane);
            v16h b0 = *(const v16h*)&WldsX[na][ks * 32 + boff];
            v16h b1 = *(const v16h*)&WldsX[na + 16][ks * 32 + boff];
            acc0 = wmma_f16(a, b0, acc0);
            acc1 = wmma_f16(a, b1, acc1);
          }
        } else {
          const float* xb = x + (size_t)arow * (LSTM_T * LSTM_I) + (size_t)s * LSTM_I + aoff;
#pragma unroll
          for (int ks = 0; ks < 8; ++ks) {
            const float* p = xb + ks * 32;
            v16h a  = load_a_f32(p, p + 8);
            v16h b0 = *(const v16h*)&WldsX[na][ks * 32 + boff];
            v16h b1 = *(const v16h*)&WldsX[na + 16][ks * 32 + boff];
            acc0 = wmma_f16(a, b0, acc0);
            acc1 = wmma_f16(a, b1, acc1);
          }
        }
      } else {
#pragma unroll
        for (int ks = 0; ks < 8; ++ks) {
          v16h a  = load_a_f16(h0in, arow, LSTM_H, ks * 32, lane);
          v16h b0 = *(const v16h*)&WldsX[na][ks * 32 + boff];
          v16h b1 = *(const v16h*)&WldsX[na + 16][ks * 32 + boff];
          acc0 = wmma_f16(a, b0, acc0);
          acc1 = wmma_f16(a, b1, acc1);
        }
      }

      // ---- h-half (serial critical path): A from own h_prev, B from registers ----
#pragma unroll
      for (int ks = 0; ks < 8; ++ks) {
        v16h a = load_a_f16(hself, arow, LSTM_H, ks * 32, lane);
        acc0 = wmma_f16(a, bregA[ks], acc0);
        acc1 = wmma_f16(a, bregB[ks], acc1);
      }

      // scatter C tiles (ISA C/D layout: VGPR r -> M = r + (lane<16?0:8))
      const int mn = lane & 15;
      const int mr = m * 16 + ((lane < 16) ? 0 : 8);
#pragma unroll
      for (int r = 0; r < 8; ++r) {
        gatesL[2 * gp][mr + r][mn]     = acc0[r];
        gatesL[2 * gp + 1][mr + r][mn] = acc1[r];
      }
      __syncthreads();

      // elementwise gate update for this h-chunk: 64x16 elems, 4 per thread
      _Float16* hout = (layer ? h1buf : h0buf) + (s & 1) * (LSTM_B * LSTM_H);
#pragma unroll
      for (int it = 0; it < 4; ++it) {
        int idx = tid * 4 + it;
        int b = idx >> 4, j = idx & 15;
        float f  = sigf(gatesL[0][b][j] + biasL[0][j]);
        float ig = sigf(gatesL[1][b][j] + biasL[1][j]);
        float o  = sigf(gatesL[2][b][j] + biasL[2][j]);
        float ch = tanh_fast(gatesL[3][b][j] + biasL[3][j]);
        float c  = f * cst[b][j] + ig * ch;
        cst[b][j] = c;
        hout[b * LSTM_H + hc * 16 + j] = (_Float16)(o * tanh_fast(c));
      }
    }
    // ---- grid barrier: release add / acquire spin on monotonic counter ----
    __syncthreads();
    if (tid == 0) {
      __hip_atomic_fetch_add(counter, 1u, __ATOMIC_RELEASE, __HIP_MEMORY_SCOPE_AGENT);
      const unsigned target = (unsigned)NWG * (unsigned)(k + 1);
      while (__hip_atomic_load(counter, __ATOMIC_ACQUIRE, __HIP_MEMORY_SCOPE_AGENT) < target)
        __builtin_amdgcn_s_sleep(1);
    }
    __syncthreads();
  }
}

// ---------------- final projection: out[64][128] = h1_final * Wout^T + bout ----------------
__global__ void __launch_bounds__(256)
lstm_proj(const _Float16* __restrict__ h1,   // [64][256] final hidden (f16)
          const float* __restrict__ Wout,    // [128][256]
          const float* __restrict__ bout,    // [128]
          float* __restrict__ out) {         // [64][128]
  const int tid = threadIdx.x, lane = tid & 31, wave = tid >> 5;
  const int m  = wave & 3;
  const int ng = wave >> 2;                  // n-tile group: tiles ng*4 .. ng*4+3
  const int arow = m * 16 + (lane & 15);
  const int mr = m * 16 + ((lane < 16) ? 0 : 8);

#pragma unroll
  for (int nt0 = 0; nt0 < 4; ++nt0) {
    const int nt = ng * 4 + nt0;
    const int ncol = nt * 16 + (lane & 15);
    v8f acc = {};
#pragma unroll
    for (int ks = 0; ks < 8; ++ks) {
      const int kb = ks * 32;
      v16h a = load_a_f16(h1, arow, LSTM_H, kb, lane);
      const float* q = Wout + (size_t)ncol * LSTM_H + kb + ((lane < 16) ? 0 : 16);
      v16h b;
#pragma unroll
      for (int j = 0; j < 16; ++j) b[j] = (_Float16)q[j];
      acc = wmma_f16(a, b, acc);
    }
    const float bo = bout[ncol];
#pragma unroll
    for (int r = 0; r < 8; ++r) out[(mr + r) * LSTM_O + ncol] = acc[r] + bo;
  }
}

// ---------------- host ----------------
extern "C" void kernel_launch(void* const* d_in, const int* in_sizes, int n_in,
                              void* d_out, int out_size, void* d_ws, size_t ws_size,
                              hipStream_t stream) {
  (void)in_sizes; (void)n_in; (void)out_size;
  const float* x    = (const float*)d_in[0];   // [64][2048][256]
  const float* Wx   = (const float*)d_in[1];   // [2][4][256][256]
  const float* Wh   = (const float*)d_in[2];   // [2][4][256][256]
  const float* bh   = (const float*)d_in[3];   // [2][4][256]
  const float* Wout = (const float*)d_in[4];   // [128][256]
  const float* bout = (const float*)d_in[5];   // [128]
  float* out = (float*)d_out;                  // [64][128]

  // workspace layout
  char* ws = (char*)d_ws;
  const size_t WCAT_OFF  = 1024;
  const size_t WCAT_SZ   = (size_t)2 * LSTM_4H * LSTM_KT * sizeof(_Float16);   // 2 MB
  const size_t HBUF_OFF  = WCAT_OFF + WCAT_SZ;
  const size_t HBUF_SZ   = (size_t)4 * LSTM_B * LSTM_H * sizeof(_Float16);     // 128 KB
  const size_t X16_OFF   = HBUF_OFF + HBUF_SZ;
  const size_t X16_SZ    = (size_t)LSTM_B * LSTM_T * LSTM_I * sizeof(_Float16);// 64 MB

  unsigned*  counter = (unsigned*)ws;
  _Float16*  wcat    = (_Float16*)(ws + WCAT_OFF);
  _Float16*  hbufs   = (_Float16*)(ws + HBUF_OFF);
  _Float16*  h0buf   = hbufs;                                  // [2][64][256]
  _Float16*  h1buf   = hbufs + 2 * LSTM_B * LSTM_H;            // [2][64][256]
  _Float16*  x16     = (ws_size >= X16_OFF + X16_SZ) ? (_Float16*)(ws + X16_OFF) : nullptr;

  lstm_convert_weights<<<1024, 256, 0, stream>>>(Wx, Wh, wcat);
  if (x16) lstm_convert_x<<<4096, 256, 0, stream>>>(x, x16);
  lstm_init_state<<<1, 256, 0, stream>>>(hbufs, counter);
  lstm_recurrent<<<NWG, 256, 0, stream>>>(x, x16, bh, wcat, h0buf, h1buf, counter);
  // final h of layer1 is at step T-1 = 2047 (odd) -> buffer index 1
  lstm_proj<<<1, 256, 0, stream>>>(h1buf + ((LSTM_T - 1) & 1) * (LSTM_B * LSTM_H),
                                   Wout, bout, out);
}
